// RNNModule_5772436046111
// MI455X (gfx1250) — compile-verified, matrix operand order checked by reading, not run on hardware
//
#include <hip/hip_runtime.h>

// ---------------------------------------------------------------------------
// 3-layer GRU (B=64, T=512, F=128, H=512) + linear head (O=128) for gfx1250.
//
//  * All intermediate tensors (XP, H) are TIME-MAJOR: row q = t*B + b.
//    -> per-timestep h block is contiguous (128 KB), scan address math is
//       constant-stride (folds into 24-bit immediate offsets).
//  * GEMMs: bf16 WMMA (v_wmma_f32_16x16x32_bf16), fp32 accumulate.
//    W slice staged once in LDS in fragment layout; each wave computes a
//    16x64 strip (4 accumulators) -> 1 A-frag conversion per 4 WMMAs.
//    Templated on N,K,MODE so all strides are compile-time.
//  * Recurrence: persistent kernel, 16 workgroups; Whh slice (96 KB bf16
//    fragments) resident in LDS; h state register-resident; grid-wide
//    atomic barrier per timestep.
// ---------------------------------------------------------------------------

typedef __attribute__((ext_vector_type(16))) __bf16 v16bf;
typedef __attribute__((ext_vector_type(8)))  float  v8f;

#define BB 64
#define TT 512
#define HH 512
#define G3 1536   // 3*H
#define MM (BB*TT)

__device__ __forceinline__ float sigm(float x) {
  return 1.0f / (1.0f + __expf(-x));
}
__device__ __forceinline__ float tanh_fast(float x) {
  return 2.0f / (1.0f + __expf(-2.0f * x)) - 1.0f;
}

// ---------------------------------------------------------------------------
// GEMM: Out[map(m)][n] = sum_k A[m][k] * W[n][k] + bias[n]
// A: [M,K] fp32 row-major; W: [N,K] fp32 row-major.
// MODE 0: map(m)=m (identity, time-major -> time-major)
// MODE 1: m = b*T+t  -> map(m) = t*B+b   (layer-0 input projection)
// MODE 2: m = t*B+b  -> map(m) = b*T+t   (head output back to [B,T,O])
// Block: 256 threads = 8 waves; tile 128(M) x 64(N); W staged in LDS.
// ---------------------------------------------------------------------------
template<int N, int K, int MODE>
__global__ void __launch_bounds__(256)
gemm_bf16_wmma(const float* __restrict__ A, const float* __restrict__ W,
               const float* __restrict__ bias, float* __restrict__ Out) {
  constexpr int KC = K / 32;
  __shared__ __bf16 ldsB[4 * KC * 32 * 16];   // [ntile][kchunk][lane][16]

  const int lane = threadIdx.x & 31;
  const int wave = threadIdx.x >> 5;
  const int lid  = lane & 15;
  const int half = lane >> 4;          // 0: k 0-7/16-23, 1: k 8-15/24-31
  const int koff = half * 8;

  const int m0 = blockIdx.x * 128 + wave * 16;
  const int n0 = blockIdx.y * 64;

  // ---- Stage W fragments (bf16) into LDS, cooperatively ------------------
  for (int s = threadIdx.x; s < 4 * KC * 32; s += 256) {
    const int ls  = s & 31;
    const int kc  = (s >> 5) % KC;
    const int nt  = (s >> 5) / KC;
    const int row = n0 + nt * 16 + (ls & 15);
    const int kb  = kc * 32 + (ls >> 4) * 8;
    const float* src = W + (size_t)row * K + kb;
    __bf16* dst = &ldsB[(size_t)s * 16];
#pragma unroll
    for (int i = 0; i < 8; ++i) {
      dst[i]     = (__bf16)src[i];
      dst[i + 8] = (__bf16)src[i + 16];
    }
  }
  __syncthreads();

  // ---- Main loop: 1 A-fragment feeds 4 WMMAs -----------------------------
  const float* __restrict__ Arow = A + (size_t)(m0 + lid) * K;
  v8f acc[4] = {};
  for (int kc = 0; kc < KC; ++kc) {
    v16bf a;
    const int kb = kc * 32 + koff;
#pragma unroll
    for (int i = 0; i < 8; ++i) {
      a[i]     = (__bf16)Arow[kb + i];
      a[i + 8] = (__bf16)Arow[kb + 16 + i];
    }
#pragma unroll
    for (int nt = 0; nt < 4; ++nt) {
      const v16bf b = *(const v16bf*)&ldsB[((nt * KC + kc) * 32 + lane) * 16];
      acc[nt] = __builtin_amdgcn_wmma_f32_16x16x32_bf16(
          false, a, false, b, (short)0, acc[nt], false, false);
    }
  }

  // ---- Store with compile-time row mapping -------------------------------
#pragma unroll
  for (int nt = 0; nt < 4; ++nt) {
    const int col = n0 + nt * 16 + lid;
    const float bv = bias[col];
#pragma unroll
    for (int r = 0; r < 8; ++r) {
      const int mrow = m0 + half * 8 + r;
      size_t orow;
      if (MODE == 0) {
        orow = (size_t)mrow;
      } else if (MODE == 1) {                 // b*T+t -> t*B+b
        orow = (size_t)(mrow & (TT - 1)) * BB + (mrow >> 9);
      } else {                                // t*B+b -> b*T+t
        orow = (size_t)(mrow & (BB - 1)) * TT + (mrow >> 6);
      }
      Out[orow * N + col] = acc[nt][r] + bv;
    }
  }
}

// ---------------------------------------------------------------------------
// Persistent GRU scan for one layer (time-major buffers).
// Grid: 16 workgroups x 256 threads (8 waves). Workgroup wg owns hidden cols
// j in [wg*32, wg*32+32). Wave w: m-tile = w&3 (16 batch rows), j-half = w>>2.
// LDS: Whh B-fragments, [6 ntile][16 kchunk][32 lane][16 bf16] = 96 KB.
// ---------------------------------------------------------------------------
__global__ void __launch_bounds__(256)
gru_scan(const float* __restrict__ XP,    // [(t*B+b), 3H] = x@Wih^T + bih
         const float* __restrict__ Whh,   // [3H, H] fp32
         const float* __restrict__ bhh,   // [3H]
         float* __restrict__ Hout,        // [(t*B+b), H]
         unsigned* __restrict__ ctr) {
  __shared__ __bf16 ldsW[6 * 16 * 32 * 16];

  const int lane = threadIdx.x & 31;
  const int wave = threadIdx.x >> 5;
  const int lid  = lane & 15;
  const int half = lane >> 4;
  const int koff = half * 8;
  const int j0   = blockIdx.x * 32;
  const unsigned NWG = gridDim.x;

  // ---- One-time: convert Whh slice fp32 -> bf16 fragments in LDS ----------
  for (int s = threadIdx.x; s < 6 * 16 * 32; s += 256) {
    const int ls = s & 31;
    const int kc = (s >> 5) & 15;
    const int nt = s >> 9;                   // 0..5 : gate = nt>>1, half = nt&1
    const int row = (nt >> 1) * HH + j0 + (nt & 1) * 16 + (ls & 15);
    const int kb  = kc * 32 + (ls >> 4) * 8;
    const float* src = Whh + (size_t)row * HH + kb;
    __bf16* dst = &ldsW[(size_t)s * 16];
#pragma unroll
    for (int i = 0; i < 8; ++i) {
      dst[i]     = (__bf16)src[i];
      dst[i + 8] = (__bf16)src[i + 16];
    }
  }
  __syncthreads();

  const int mt  = wave & 3;                  // batch tile (16 rows)
  const int jt  = wave >> 2;                 // which 16-col half of the slice
  const int myj = j0 + jt * 16 + lid;        // this lane's hidden column
  const int b8  = mt * 16 + half * 8;        // first batch row of this lane

  const float b_r = bhh[myj];
  const float b_z = bhh[HH + myj];
  const float b_n = bhh[2 * HH + myj];

  v8f hreg = {};                             // h_{t-1} for this lane's 8 (b,j)

  // Incrementally-updated bases (time-major: row q = t*B + b)
  size_t xbase = (size_t)b8 * G3 + myj;              // XP[(t*B + b8)*G3 + myj]
  size_t hbase = (size_t)b8 * HH + myj;              // Hout[(t*B + b8)*H + myj]
  const float* __restrict__ Abase =
      Hout + (size_t)(mt * 16 + lid) * HH;           // + (t-1)*B*H per step

  for (int t = 0; t < TT; ++t) {
    v8f c[3] = {};                           // gh accumulators: r, z, n

    if (t > 0) {
      const float* __restrict__ Ap = Abase + (size_t)(t - 1) * (BB * HH);
      for (int kc = 0; kc < 16; ++kc) {
        v16bf a;
        const int kb = kc * 32 + koff;
#pragma unroll
        for (int i = 0; i < 8; ++i) {
          a[i]     = (__bf16)Ap[kb + i];
          a[i + 8] = (__bf16)Ap[kb + 16 + i];
        }
#pragma unroll
        for (int g = 0; g < 3; ++g) {
          const v16bf b = *(const v16bf*)
              &ldsW[((((2 * g + jt) * 16 + kc) * 32) + lane) * 16];
          c[g] = __builtin_amdgcn_wmma_f32_16x16x32_bf16(
              false, a, false, b, (short)0, c[g], false, false);
        }
      }
    }

    // ---- Gate math + state update (fp32); constant per-r strides --------
#pragma unroll
    for (int r = 0; r < 8; ++r) {
      const float xr = XP[xbase + r * G3];
      const float xz = XP[xbase + r * G3 + HH];
      const float xn = XP[xbase + r * G3 + 2 * HH];
      const float gr = sigm(xr + c[0][r] + b_r);
      const float gz = sigm(xz + c[1][r] + b_z);
      const float gn = tanh_fast(xn + gr * (c[2][r] + b_n));
      const float h  = (1.0f - gz) * gn + gz * hreg[r];
      hreg[r] = h;
      Hout[hbase + r * HH] = h;
    }
    xbase += (size_t)BB * G3;
    hbase += (size_t)BB * HH;

    // ---- Grid-wide barrier: all 16 WGs align on step t ------------------
    __threadfence();
    __syncthreads();
    if (threadIdx.x == 0) {
      atomicAdd(ctr, 1u);
      const unsigned target = (unsigned)(t + 1) * NWG;
      while (__hip_atomic_load(ctr, __ATOMIC_RELAXED,
                               __HIP_MEMORY_SCOPE_AGENT) < target) {
        __builtin_amdgcn_s_sleep(1);
      }
    }
    __syncthreads();
    __threadfence();
  }
}

// ---------------------------------------------------------------------------
extern "C" void kernel_launch(void* const* d_in, const int* in_sizes, int n_in,
                              void* d_out, int out_size, void* d_ws, size_t ws_size,
                              hipStream_t stream) {
  const float* x     = (const float*)d_in[0];
  const float* Wih0  = (const float*)d_in[1];
  const float* Whh0  = (const float*)d_in[2];
  const float* bih0  = (const float*)d_in[3];
  const float* bhh0  = (const float*)d_in[4];
  const float* Wih1  = (const float*)d_in[5];
  const float* Whh1  = (const float*)d_in[6];
  const float* bih1  = (const float*)d_in[7];
  const float* bhh1  = (const float*)d_in[8];
  const float* Wih2  = (const float*)d_in[9];
  const float* Whh2  = (const float*)d_in[10];
  const float* bih2  = (const float*)d_in[11];
  const float* bhh2  = (const float*)d_in[12];
  const float* Wout  = (const float*)d_in[13];
  const float* bout  = (const float*)d_in[14];
  float* out = (float*)d_out;

  char* ws = (char*)d_ws;
  const size_t XP_BYTES = (size_t)MM * G3 * 4;   // 201.3 MB
  const size_t H_BYTES  = (size_t)MM * HH * 4;   //  67.1 MB
  float*    XP  = (float*)ws;
  float*    HA  = (float*)(ws + XP_BYTES);
  float*    HB  = (float*)(ws + XP_BYTES + H_BYTES);
  unsigned* ctr = (unsigned*)(ws + XP_BYTES + 2 * H_BYTES);

  const dim3 blk(256);
  const dim3 gproj(MM / 128, G3 / 64);
  const dim3 ghead(MM / 128, 128 / 64);
  const dim3 gscan(HH / 32);   // 16 persistent workgroups

  // Layer 0: x is [B,T,F] (rows b*T+t) -> XP time-major (MODE 1)
  gemm_bf16_wmma<G3, 128, 1><<<gproj, blk, 0, stream>>>(x, Wih0, bih0, XP);
  hipMemsetAsync(ctr, 0, sizeof(unsigned), stream);
  gru_scan<<<gscan, blk, 0, stream>>>(XP, Whh0, bhh0, HA, ctr);

  // Layer 1: time-major -> time-major (MODE 0)
  gemm_bf16_wmma<G3, HH, 0><<<gproj, blk, 0, stream>>>(HA, Wih1, bih1, XP);
  hipMemsetAsync(ctr, 0, sizeof(unsigned), stream);
  gru_scan<<<gscan, blk, 0, stream>>>(XP, Whh1, bhh1, HB, ctr);

  // Layer 2
  gemm_bf16_wmma<G3, HH, 0><<<gproj, blk, 0, stream>>>(HB, Wih2, bih2, XP);
  hipMemsetAsync(ctr, 0, sizeof(unsigned), stream);
  gru_scan<<<gscan, blk, 0, stream>>>(XP, Whh2, bhh2, HA, ctr);

  // Head: time-major -> [B,T,O] (MODE 2)
  gemm_bf16_wmma<128, HH, 2><<<ghead, blk, 0, stream>>>(HA, Wout, bout, out);
}
